// Diffusion_8950711845685
// MI455X (gfx1250) — matrix operand order repaired and checked by reading, not compiled, not run on hardware
//
#include <hip/hip_runtime.h>
#include <math.h>

#define B_TOT 4096
#define OBS   64
#define ACT   32
#define HDIM  256
#define TDIM  16
#define TSTEPS 32
#define INDIM 112     // ACT + TDIM + OBS
#define KP1   128     // layer-1 K padded to 128
#define STRIDE 264    // LDS activation row stride in halves (bank-conflict-free)
#define MROWS 32      // rows per block (2 M-tiles)

typedef __attribute__((ext_vector_type(16))) _Float16 v16h;
typedef __attribute__((ext_vector_type(8)))  _Float16 v8h;
typedef __attribute__((ext_vector_type(8)))  float    v8f;

// ---------------- helpers ----------------

// mish(x) = x * tanh(softplus(x)).  With u = e^x:
//   tanh(log1p(u)) = ((1+u)^2 - 1) / ((1+u)^2 + 1) = u(u+2) / (u(u+2)+2)   (exact)
// Clamp x<=22 so w stays finite; there the ratio rounds to 1.0 and mish(x)==x.
__device__ __forceinline__ float mishf(float x) {
    float u = __expf(fminf(x, 22.0f));
    float w = u * (u + 2.0f);
    return x * w * __builtin_amdgcn_rcpf(w + 2.0f);
}

__device__ __forceinline__ float gaussf(unsigned long long s) {
    // splitmix64 -> Box-Muller (deterministic; stands in for jax threefry)
    unsigned long long z = s + 0x9E3779B97F4A7C15ULL;
    z = (z ^ (z >> 30)) * 0xBF58476D1CE4E5B9ULL;
    z = (z ^ (z >> 27)) * 0x94D049BB133111EBULL;
    z ^= (z >> 31);
    unsigned u1 = (unsigned)z;
    unsigned u2 = (unsigned)(z >> 32);
    float f1 = ((float)u1 + 1.0f) * 2.3283064365386963e-10f;  // (0,1]
    float f2 = (float)u2 * 2.3283064365386963e-10f;
    return sqrtf(-2.0f * __logf(f1)) * __cosf(6.28318530717958647f * f2);
}

// A-fragment (16x32 f16, M x K) per ISA 7.12.2:
// lanes 0-15: halves0-7 = K 0..7,  halves8-15 = K 16..23   (M = lane)
// lanes16-31: halves0-7 = K 8..15, halves8-15 = K 24..31   (M = lane-16)
__device__ __forceinline__ v16h load_afrag(const _Float16* act, int lane, int kk) {
    int m  = lane & 15;
    int kh = (lane >> 4) * 8;
    const _Float16* p = act + m * STRIDE + kk * 32 + kh;
    v8h lo = *(const v8h*)p;
    v8h hi = *(const v8h*)(p + 16);
    v16h a;
#pragma unroll
    for (int i = 0; i < 8; ++i) { a[i] = lo[i]; a[8 + i] = hi[i]; }
    return a;
}

// B-fragment (32x16 f16, K x N), weights pre-transposed to [N][Kpad]; same K order as A:
// lanes 0-15 -> K 0..7,16..23; lanes 16-31 -> K 8..15,24..31.
__device__ __forceinline__ v16h load_bfrag(const _Float16* WT, int lane, int kk,
                                           int nbase, int kp) {
    int n  = nbase + (lane & 15);
    int ko = kk * 32 + (lane >> 4) * 8;
    const _Float16* p = WT + n * kp + ko;
    v8h lo = *(const v8h*)p;
    v8h hi = *(const v8h*)(p + 16);
    v16h b;
#pragma unroll
    for (int i = 0; i < 8; ++i) { b[i] = lo[i]; b[8 + i] = hi[i]; }
    return b;
}

__device__ __forceinline__ void store_ctile(_Float16* nxt, const v8f& acc,
                                            int mbase, int n, float bn, bool do_mish) {
#pragma unroll
    for (int r = 0; r < 8; ++r) {
        float v = acc[r] + bn;
        if (do_mish) v = mishf(v);
        nxt[(mbase + r) * STRIDE + n] = (_Float16)v;
    }
}

// One fused hidden layer over a 32-row block: 2 M-tiles x 2 N-tiles per wave.
// Each global B-fragment is reused by both M-tiles.
__device__ __forceinline__ void gemm_layer(const _Float16* act, _Float16* nxt,
                                           const _Float16* __restrict__ WT,
                                           const float* __restrict__ bias,
                                           int kp, int wave, int lane, bool do_mish) {
    int nb0 = wave * 32;
    v8f acc00 = {}, acc01 = {}, acc10 = {}, acc11 = {};
    int ksteps = kp >> 5;
    for (int kk = 0; kk < ksteps; ++kk) {
        v16h a0 = load_afrag(act, lane, kk);
        v16h a1 = load_afrag(act + 16 * STRIDE, lane, kk);
        v16h b0 = load_bfrag(WT, lane, kk, nb0, kp);
        v16h b1 = load_bfrag(WT, lane, kk, nb0 + 16, kp);
        acc00 = __builtin_amdgcn_wmma_f32_16x16x32_f16(false, a0, false, b0, (short)0, acc00, false, false);
        acc01 = __builtin_amdgcn_wmma_f32_16x16x32_f16(false, a0, false, b1, (short)0, acc01, false, false);
        acc10 = __builtin_amdgcn_wmma_f32_16x16x32_f16(false, a1, false, b0, (short)0, acc10, false, false);
        acc11 = __builtin_amdgcn_wmma_f32_16x16x32_f16(false, a1, false, b1, (short)0, acc11, false, false);
    }
    // C layout: VGPR r -> M = r + 8*(lane>>4), N = nbase + (lane&15)
    int m0 = (lane >> 4) * 8;
    int nl = lane & 15;
    float bn0 = bias[nb0 + nl];
    float bn1 = bias[nb0 + 16 + nl];
    store_ctile(nxt, acc00, m0,      nb0 + nl,      bn0, do_mish);
    store_ctile(nxt, acc01, m0,      nb0 + 16 + nl, bn1, do_mish);
    store_ctile(nxt, acc10, m0 + 16, nb0 + nl,      bn0, do_mish);
    store_ctile(nxt, acc11, m0 + 16, nb0 + 16 + nl, bn1, do_mish);
}

// ---------------- prep kernels ----------------

// dst[n*KP + k] = (k < K) ? (f16)src[k*N + n] : 0   (transpose + f32->f16)
__global__ void cvt_transpose(const float* __restrict__ src, _Float16* __restrict__ dst,
                              int K, int N, int KP) {
    int i = blockIdx.x * blockDim.x + threadIdx.x;
    if (i >= N * KP) return;
    int n = i / KP;
    int k = i - n * KP;
    dst[i] = (k < K) ? (_Float16)src[k * N + n] : (_Float16)0.0f;
}

// Precompute time-embedding MLP output for all T steps: TE[T][16].
__global__ void te_kernel(const float* __restrict__ Wt1, const float* __restrict__ bt1,
                          const float* __restrict__ Wt2, const float* __restrict__ bt2,
                          float* __restrict__ TE) {
    int i = threadIdx.x;
    if (i >= TSTEPS) return;
    float emb[TDIM];
    float neg = -__logf(10000.0f) / 7.0f;  // half-1 = 7
#pragma unroll
    for (int j = 0; j < 8; ++j) {
        float ang = (float)i * __expf((float)j * neg);
        emb[j]     = __sinf(ang);
        emb[8 + j] = __cosf(ang);
    }
    float h[2 * TDIM];
    for (int o = 0; o < 2 * TDIM; ++o) {
        float s = bt1[o];
        for (int k = 0; k < TDIM; ++k) s += emb[k] * Wt1[k * (2 * TDIM) + o];
        h[o] = mishf(s);
    }
    for (int o = 0; o < TDIM; ++o) {
        float s = bt2[o];
        for (int k = 0; k < 2 * TDIM; ++k) s += h[k] * Wt2[k * TDIM + o];
        TE[i * TDIM + o] = s;
    }
}

__global__ void init_x0(float* __restrict__ xbuf, int n) {
    int i = blockIdx.x * blockDim.x + threadIdx.x;
    if (i < n) xbuf[i] = gaussf(0xD1FF0000ABCD1234ULL + (unsigned long long)i);
}

// ---------------- fused per-step MLP + diffusion update ----------------

__global__ __launch_bounds__(256) void step_kernel(
    const float* __restrict__ state,
    const float* __restrict__ b1, const float* __restrict__ b2,
    const float* __restrict__ b3, const float* __restrict__ bf,
    const _Float16* __restrict__ WT1, const _Float16* __restrict__ WT2,
    const _Float16* __restrict__ WT3, const _Float16* __restrict__ WTf,
    const float* __restrict__ TE, float* __restrict__ xbuf,
    float* __restrict__ out,
    int step, float a1, float a2, float c1, float c2, float sigma, int is_last)
{
    __shared__ __align__(32) _Float16 actA[MROWS * STRIDE];
    __shared__ __align__(32) _Float16 actB[MROWS * STRIDE];
    __shared__ float epsb[MROWS * ACT];

    int tid  = threadIdx.x;
    int lane = tid & 31;
    int wave = tid >> 5;
    int row0 = blockIdx.x * MROWS;
    const float* te = TE + step * TDIM;

    // Stage 0: assemble layer-1 input concat[x, te, state] -> LDS f16 (padded to 128)
    for (int idx = tid; idx < MROWS * KP1; idx += 256) {
        int m = idx >> 7;
        int k = idx & 127;
        float v;
        if (k < ACT)                v = xbuf[(row0 + m) * ACT + k];
        else if (k < ACT + TDIM)    v = te[k - ACT];
        else if (k < INDIM)         v = state[(row0 + m) * OBS + (k - ACT - TDIM)];
        else                        v = 0.0f;
        actA[m * STRIDE + k] = (_Float16)v;
    }
    __syncthreads();

    gemm_layer(actA, actB, WT1, b1, KP1,  wave, lane, true);
    __syncthreads();
    gemm_layer(actB, actA, WT2, b2, HDIM, wave, lane, true);
    __syncthreads();
    gemm_layer(actA, actB, WT3, b3, HDIM, wave, lane, true);
    __syncthreads();

    // Final layer: Nout = 32 -> two N-tiles handled by waves 0 and 1 (wave-uniform
    // branch, EXEC all-ones within the wave). Each wave does both M-tiles.
    if (wave < 2) {
        int nb = wave * 16;
        v8f acc0 = {}, acc1 = {};
        for (int kk = 0; kk < HDIM / 32; ++kk) {
            v16h b  = load_bfrag(WTf, lane, kk, nb, HDIM);
            v16h a0 = load_afrag(actB, lane, kk);
            v16h a1 = load_afrag(actB + 16 * STRIDE, lane, kk);
            acc0 = __builtin_amdgcn_wmma_f32_16x16x32_f16(false, a0, false, b, (short)0, acc0, false, false);
            acc1 = __builtin_amdgcn_wmma_f32_16x16x32_f16(false, a1, false, b, (short)0, acc1, false, false);
        }
        int m0 = (lane >> 4) * 8;
        int n  = nb + (lane & 15);
        float bn = bf[n];
#pragma unroll
        for (int r = 0; r < 8; ++r) {
            epsb[(m0 + r) * ACT + n]      = acc0[r] + bn;
            epsb[(m0 + 16 + r) * ACT + n] = acc1[r] + bn;
        }
    }
    __syncthreads();

    // Diffusion posterior update (1024 elements, 4 per thread)
#pragma unroll
    for (int e = tid; e < MROWS * ACT; e += 256) {
        int m  = e >> 5;
        int cc = e & 31;
        int row = row0 + m;
        float x   = xbuf[row * ACT + cc];
        float eps = epsb[m * ACT + cc];
        float xr  = fminf(fmaxf(a1 * x - a2 * eps, -1.0f), 1.0f);
        float val = c1 * xr + c2 * x;
        if (sigma > 0.0f) {
            unsigned long long seed =
                ((unsigned long long)(step + 1) << 40) ^ (unsigned long long)(row * ACT + cc);
            val += sigma * gaussf(seed);
        }
        if (is_last) out[row * ACT + cc] = fminf(fmaxf(val, -1.0f), 1.0f);
        else         xbuf[row * ACT + cc] = val;
    }
}

// ---------------- host ----------------

extern "C" void kernel_launch(void* const* d_in, const int* in_sizes, int n_in,
                              void* d_out, int out_size, void* d_ws, size_t ws_size,
                              hipStream_t stream) {
    const float* state = (const float*)d_in[0];
    const float* Wt1   = (const float*)d_in[1];
    const float* bt1   = (const float*)d_in[2];
    const float* Wt2   = (const float*)d_in[3];
    const float* bt2   = (const float*)d_in[4];
    const float* W1    = (const float*)d_in[5];
    const float* b1    = (const float*)d_in[6];
    const float* W2    = (const float*)d_in[7];
    const float* b2    = (const float*)d_in[8];
    const float* W3    = (const float*)d_in[9];
    const float* b3    = (const float*)d_in[10];
    const float* Wf    = (const float*)d_in[11];
    const float* bf    = (const float*)d_in[12];

    char* ws = (char*)d_ws;
    float*    xbuf = (float*)(ws + 0);          // 4096*32*4   = 524288
    float*    TE   = (float*)(ws + 524288);     // 32*16*4     = 2048
    _Float16* WT1h = (_Float16*)(ws + 526336);  // 256*128*2   = 65536
    _Float16* WT2h = (_Float16*)(ws + 591872);  // 256*256*2   = 131072
    _Float16* WT3h = (_Float16*)(ws + 722944);  // 256*256*2   = 131072
    _Float16* WTfh = (_Float16*)(ws + 854016);  // 32*256*2    = 16384

    // One-time (per launch) prep: weight transpose/convert, TE table, x0 init.
    cvt_transpose<<<(HDIM * KP1 + 255) / 256, 256, 0, stream>>>(W1, WT1h, INDIM, HDIM, KP1);
    cvt_transpose<<<(HDIM * HDIM + 255) / 256, 256, 0, stream>>>(W2, WT2h, HDIM, HDIM, HDIM);
    cvt_transpose<<<(HDIM * HDIM + 255) / 256, 256, 0, stream>>>(W3, WT3h, HDIM, HDIM, HDIM);
    cvt_transpose<<<(ACT * HDIM + 255) / 256, 256, 0, stream>>>(Wf, WTfh, HDIM, ACT, HDIM);
    te_kernel<<<1, 32, 0, stream>>>(Wt1, bt1, Wt2, bt2, TE);
    init_x0<<<(B_TOT * ACT + 255) / 256, 256, 0, stream>>>(xbuf, B_TOT * ACT);

    // DDPM schedule on host (deterministic scalars baked into each launch).
    float betas[TSTEPS], alphas[TSTEPS], ac[TSTEPS], acp[TSTEPS];
    float prod = 1.0f;
    for (int i = 0; i < TSTEPS; ++i) {
        betas[i]  = 1e-4f + (0.02f - 1e-4f) * (float)i / (float)(TSTEPS - 1);
        alphas[i] = 1.0f - betas[i];
        prod *= alphas[i];
        ac[i]  = prod;
        acp[i] = (i == 0) ? 1.0f : ac[i - 1];
    }

    for (int i = TSTEPS - 1; i >= 0; --i) {
        float a1 = sqrtf(1.0f / ac[i]);
        float a2 = sqrtf(1.0f / ac[i] - 1.0f);
        float pv = betas[i] * (1.0f - acp[i]) / (1.0f - ac[i]);
        float lv = logf(fmaxf(pv, 1e-20f));
        float c1 = betas[i] * sqrtf(acp[i]) / (1.0f - ac[i]);
        float c2 = (1.0f - acp[i]) * sqrtf(alphas[i]) / (1.0f - ac[i]);
        float sigma = (i > 0) ? expf(0.5f * lv) : 0.0f;
        step_kernel<<<B_TOT / MROWS, 256, 0, stream>>>(
            state, b1, b2, b3, bf, WT1h, WT2h, WT3h, WTfh,
            TE, xbuf, (float*)d_out, i, a1, a2, c1, c2, sigma, (i == 0) ? 1 : 0);
    }
}
